// HybridGraphAttention_8615704396338
// MI455X (gfx1250) — matrix operand order repaired and checked by reading, not compiled, not run on hardware
//
#include <hip/hip_runtime.h>

// ---------------- problem constants ----------------
constexpr int   Nn   = 50000;
constexpr int   Ne   = 800000;
constexpr int   INd  = 128;
constexpr int   EDd  = 32;
constexpr int   Hh   = 4;
constexpr int   Cc   = 32;
constexpr int   HC   = 128;
constexpr float SCALE = 0.17677669529663688f;  // 1/sqrt(32)

typedef __attribute__((ext_vector_type(16))) __bf16 v16bf;
typedef __attribute__((ext_vector_type(8)))  float  v8f;

#define DEVFN __device__ __forceinline__

DEVFN unsigned short f2bfbits(float f) {
  unsigned u = __float_as_uint(f);
  u += 0x7FFFu + ((u >> 16) & 1u);          // round-to-nearest-even
  return (unsigned short)(u >> 16);
}
DEVFN float  bf2f(unsigned short s) { return __uint_as_float(((unsigned)s) << 16); }
DEVFN __bf16 bfraw(unsigned short s) { return __builtin_bit_cast(__bf16, s); }
// order-preserving float<->uint map for atomicMax-based segment max
DEVFN unsigned fenc(float f) { unsigned u = __float_as_uint(f); return (u & 0x80000000u) ? ~u : (u | 0x80000000u); }
DEVFN float    fdec(unsigned u) { return __uint_as_float((u & 0x80000000u) ? (u & 0x7FFFFFFFu) : ~u); }

// ---------------- f32 -> bf16 (optionally strided/offset source) ----------------
__global__ __launch_bounds__(256)
void k_cvt_bf16(const float* __restrict__ src, unsigned short* __restrict__ dst,
                int cols, int ld, int colOff, long long total) {
  long long i = (long long)blockIdx.x * 256 + threadIdx.x;
  if (i >= total) return;
  int r = (int)(i / cols);
  int c = (int)(i - (long long)r * cols);
  dst[i] = f2bfbits(src[(long long)r * ld + colOff + c]);
}

// ---------------- WMMA GEMM: D[M,128] = A[M,128] * B[128,128]^T (+Cacc)(+bias) ----------------
// one block = one 16-row tile of A; 8 waves cover the 8 N-tiles of the 128-wide output.
__global__ __launch_bounds__(256)
void k_wmma_gemm128(const unsigned short* __restrict__ A,
                    const unsigned short* __restrict__ B,
                    const float* __restrict__ Cacc,
                    const float* __restrict__ bias,
                    float* __restrict__ Df32,
                    unsigned short* __restrict__ Dbf16) {
  __shared__ unsigned short lA[16 * 128];
  const int row0 = blockIdx.x * 16;
  // coalesced stage of the 16x128 bf16 A tile into LDS
  for (int i = threadIdx.x; i < 16 * 128; i += 256)
    lA[i] = A[(size_t)row0 * 128 + i];
  __syncthreads();

  const int lane  = threadIdx.x & 31;
  const int ntile = threadIdx.x >> 5;      // 0..7 -> N-tile
  const int half  = lane >> 4;             // lane group (K split)
  const int nloc  = lane & 15;
  const int arow  = lane & 15;             // A: lanes 0-15 and 16-31 both map to rows 0-15

  v8f acc = {};
  #pragma unroll
  for (int kb = 0; kb < 128; kb += 32) {
    v16bf a, b;
    // A 16x32 bf16, this code's contiguous K order (not the ISA table order): lanes 0-15 hold K 0..15, lanes 16-31 hold K 16..31
    #pragma unroll
    for (int i = 0; i < 16; ++i)
      a[i] = bfraw(lA[arow * 128 + kb + half * 16 + i]);
    // B 32x16 bf16, same contiguous K order as A: lane n holds column n; lanes 0-15 K=0..15, lanes 16-31 K=16..31
    const unsigned short* brow = B + (size_t)(ntile * 16 + nloc) * 128 + kb + half * 16;
    #pragma unroll
    for (int i = 0; i < 16; ++i)
      b[i] = bfraw(brow[i]);
    acc = __builtin_amdgcn_wmma_f32_16x16x32_bf16(false, a, false, b, (short)0, acc, false, false);
  }

  const int ncol = ntile * 16 + nloc;
  #pragma unroll
  for (int r = 0; r < 8; ++r) {            // C/D: VGPR r -> M = r + 8*half, N = lane&15
    size_t o = (size_t)(row0 + r + half * 8) * 128 + ncol;
    float v = acc[r];
    if (Cacc)  v += Cacc[o];
    if (bias)  v += bias[ncol];
    if (Df32)  Df32[o] = v;
    if (Dbf16) Dbf16[o] = f2bfbits(v);
  }
}

// ---------------- edge attention scores (3 branches) + segment max ----------------
DEVFN float branch_score(const unsigned short* __restrict__ Q,
                         const unsigned short* __restrict__ Kp,
                         const float* w, const float* d,
                         long long qo, long long ko, int h) {
  float t = 0.f;
  #pragma unroll 4
  for (int c = 0; c < 32; ++c) {
    const float* wr = w + (h * 32 + c) * 32;
    float kr = 0.f;
    #pragma unroll
    for (int j = 0; j < 32; ++j) kr = fmaf(d[j], wr[j], kr);
    t = fmaf(bf2f(Q[qo + c]), bf2f(Kp[ko + c]) + kr, t);
  }
  return t * SCALE;
}

__global__ __launch_bounds__(256)
void k_edge_scores(const int* __restrict__ ei, const float* __restrict__ eattr,
                   const unsigned short* __restrict__ Q1, const unsigned short* __restrict__ K1,
                   const unsigned short* __restrict__ Q2, const unsigned short* __restrict__ K2,
                   const unsigned short* __restrict__ Qa, const unsigned short* __restrict__ Ka,
                   const float* __restrict__ Wkr1, const float* __restrict__ Wkr2,
                   const float* __restrict__ Wkra,
                   float* __restrict__ t1, float* __restrict__ t2, float* __restrict__ ta,
                   unsigned* __restrict__ m1, unsigned* __restrict__ m2, unsigned* __restrict__ ma) {
  __shared__ float w1[HC * EDd], w2[HC * EDd], wa[HC * EDd];  // 48 KB of 320 KB LDS
  for (int i = threadIdx.x; i < HC * EDd; i += 256) { w1[i] = Wkr1[i]; w2[i] = Wkr2[i]; wa[i] = Wkra[i]; }
  __syncthreads();
  long long tid = (long long)blockIdx.x * 256 + threadIdx.x;
  if (tid >= (long long)Ne * Hh) return;
  int e = (int)(tid >> 2), h = (int)(tid & 3);
  int src = ei[e], dst = ei[Ne + e];
  float d[32];
  const float* ep = eattr + (long long)e * EDd;
  #pragma unroll
  for (int j = 0; j < 32; ++j) d[j] = ep[j];
  long long qo = (long long)dst * HC + h * Cc;
  long long ko = (long long)src * HC + h * Cc;
  float s1 = branch_score(Q1, K1, w1, d, qo, ko, h);
  float s2 = branch_score(Q2, K2, w2, d, qo, ko, h);
  float sa = branch_score(Qa, Ka, wa, d, qo, ko, h);
  t1[tid] = s1; t2[tid] = s2; ta[tid] = sa;
  int ni = dst * Hh + h;
  atomicMax(&m1[ni], fenc(s1));
  atomicMax(&m2[ni], fenc(s2));
  atomicMax(&ma[ni], fenc(sa));
}

// ---------------- exp(t - m) + segment sum ----------------
__global__ __launch_bounds__(256)
void k_edge_exp(const int* __restrict__ ei,
                float* __restrict__ t1, float* __restrict__ t2, float* __restrict__ ta,
                const unsigned* __restrict__ m1, const unsigned* __restrict__ m2,
                const unsigned* __restrict__ ma,
                float* __restrict__ z1, float* __restrict__ z2, float* __restrict__ za) {
  long long tid = (long long)blockIdx.x * 256 + threadIdx.x;
  if (tid >= (long long)Ne * Hh) return;
  int e = (int)(tid >> 2), h = (int)(tid & 3);
  int ni = ei[Ne + e] * Hh + h;
  float x1 = __expf(t1[tid] - fdec(m1[ni])); t1[tid] = x1; atomicAdd(&z1[ni], x1);
  float x2 = __expf(t2[tid] - fdec(m2[ni])); t2[tid] = x2; atomicAdd(&z2[ni], x2);
  float xa = __expf(ta[tid] - fdec(ma[ni])); ta[tid] = xa; atomicAdd(&za[ni], xa);
}

// ---------------- weighted message scatter (both branches) ----------------
__global__ __launch_bounds__(256)
void k_edge_msg(const int* __restrict__ ei, const float* __restrict__ eattr,
                const unsigned short* __restrict__ Vd, const unsigned short* __restrict__ Va,
                const float* __restrict__ Wvrd, const float* __restrict__ Wvra,
                const float* __restrict__ t1, const float* __restrict__ t2,
                const float* __restrict__ ta,
                const float* __restrict__ z1, const float* __restrict__ z2,
                const float* __restrict__ za,
                const float* __restrict__ lamp,
                float* __restrict__ hagg, float* __restrict__ hact) {
  __shared__ float wd[HC * EDd], wv[HC * EDd];  // 32 KB
  for (int i = threadIdx.x; i < HC * EDd; i += 256) { wd[i] = Wvrd[i]; wv[i] = Wvra[i]; }
  __syncthreads();
  long long tid = (long long)blockIdx.x * 256 + threadIdx.x;
  if (tid >= (long long)Ne * Hh) return;
  int e = (int)(tid >> 2), h = (int)(tid & 3);
  int src = ei[e], dst = ei[Ne + e];
  int ni = dst * Hh + h;
  float lam = lamp[0];
  float ad = t1[tid] / (z1[ni] + 1e-16f) - lam * (t2[tid] / (z2[ni] + 1e-16f));
  float aa = ta[tid] / (za[ni] + 1e-16f);
  float d[32];
  const float* ep = eattr + (long long)e * EDd;
  #pragma unroll
  for (int j = 0; j < 32; ++j) d[j] = ep[j];
  long long vo = (long long)src * HC + h * Cc;
  long long oo = (long long)dst * HC + h * Cc;
  #pragma unroll 2
  for (int c = 0; c < 32; ++c) {
    const float* rd = wd + (h * 32 + c) * 32;
    const float* rv = wv + (h * 32 + c) * 32;
    float vrd = 0.f, vra = 0.f;
    #pragma unroll
    for (int j = 0; j < 32; ++j) { vrd = fmaf(d[j], rd[j], vrd); vra = fmaf(d[j], rv[j], vra); }
    atomicAdd(&hagg[oo + c], (bf2f(Vd[vo + c]) + vrd) * ad);
    atomicAdd(&hact[oo + c], (bf2f(Va[vo + c]) + vra) * aa);
  }
}

// ---------------- residual + RMSNorm ----------------
__global__ __launch_bounds__(128)
void k_fuse(const float* __restrict__ proj, const float* __restrict__ xinp,
            const float* __restrict__ rmsw, float* __restrict__ out) {
  __shared__ float red[128];
  int i = blockIdx.x, j = threadIdx.x;
  size_t o = (size_t)i * 128 + j;
  float v = proj[o] + xinp[o];
  red[j] = v * v;
  __syncthreads();
  #pragma unroll
  for (int s = 64; s > 0; s >>= 1) { if (j < s) red[j] += red[j + s]; __syncthreads(); }
  float inv = rsqrtf(red[0] * (1.0f / 128.0f) + 1e-6f);
  out[o] = v * inv * rmsw[j];
}

// ---------------- host orchestration ----------------
extern "C" void kernel_launch(void* const* d_in, const int* in_sizes, int n_in,
                              void* d_out, int out_size, void* d_ws, size_t ws_size,
                              hipStream_t stream) {
  (void)in_sizes; (void)n_in; (void)out_size; (void)ws_size;
  const float* x    = (const float*)d_in[0];
  const int*   ei   = (const int*)  d_in[1];
  const float* ea   = (const float*)d_in[2];
  const float* Wq1  = (const float*)d_in[3];
  const float* Wq2  = (const float*)d_in[4];
  const float* Wk1  = (const float*)d_in[5];
  const float* Wk2  = (const float*)d_in[6];
  const float* Wvd  = (const float*)d_in[7];
  const float* Wkr1 = (const float*)d_in[8];
  const float* Wkr2 = (const float*)d_in[9];
  const float* Wvrd = (const float*)d_in[10];
  const float* Woutd= (const float*)d_in[11];
  const float* lam  = (const float*)d_in[12];
  const float* Wqa  = (const float*)d_in[13];
  const float* Wka  = (const float*)d_in[14];
  const float* Wva  = (const float*)d_in[15];
  const float* Wkra = (const float*)d_in[16];
  const float* Wvra = (const float*)d_in[17];
  const float* Wouta= (const float*)d_in[18];
  const float* Wproj= (const float*)d_in[19];
  const float* bproj= (const float*)d_in[20];
  const float* Win  = (const float*)d_in[21];
  const float* b_in = (const float*)d_in[22];
  const float* rmsw = (const float*)d_in[23];
  float* out = (float*)d_out;

  // bump allocator over workspace
  char* wsb = (char*)d_ws;
  size_t off = 0;
  auto alloc = [&](size_t bytes) -> char* {
    off = (off + 255) & ~(size_t)255;
    char* p = wsb + off;
    off += bytes;
    return p;
  };
  const size_t NHC = (size_t)Nn * HC;
  const size_t EH  = (size_t)Ne * Hh;
  const size_t NH  = (size_t)Nn * Hh;

  unsigned short* xbf = (unsigned short*)alloc((size_t)Nn * INd * 2);
  unsigned short* wB[13];
  for (int i = 0; i < 13; ++i) wB[i] = (unsigned short*)alloc(16384 * 2);
  // wB order: 0 Wq1,1 Wq2,2 Wk1,3 Wk2,4 Wvd,5 Wqa,6 Wka,7 Wva,8 Win,9 Woutd,10 Wouta,11 Wp1,12 Wp2
  unsigned short* q1p = (unsigned short*)alloc(NHC * 2);
  unsigned short* q2p = (unsigned short*)alloc(NHC * 2);
  unsigned short* k1p = (unsigned short*)alloc(NHC * 2);
  unsigned short* k2p = (unsigned short*)alloc(NHC * 2);
  unsigned short* vdp = (unsigned short*)alloc(NHC * 2);
  unsigned short* qap = (unsigned short*)alloc(NHC * 2);
  unsigned short* kap = (unsigned short*)alloc(NHC * 2);
  unsigned short* vap = (unsigned short*)alloc(NHC * 2);
  float*    xinp = (float*)alloc(NHC * 4);
  float*    t1   = (float*)alloc(EH * 4);
  float*    t2   = (float*)alloc(EH * 4);
  float*    ta   = (float*)alloc(EH * 4);
  unsigned* m1   = (unsigned*)alloc(NH * 4);
  unsigned* m2   = (unsigned*)alloc(NH * 4);
  unsigned* ma   = (unsigned*)alloc(NH * 4);
  float*    z1   = (float*)alloc(NH * 4);
  float*    z2   = (float*)alloc(NH * 4);
  float*    za   = (float*)alloc(NH * 4);
  float*    hagg = (float*)alloc(NHC * 4);
  float*    hact = (float*)alloc(NHC * 4);
  unsigned short* habf = (unsigned short*)alloc(NHC * 2);
  unsigned short* hvbf = (unsigned short*)alloc(NHC * 2);
  unsigned short* g1   = (unsigned short*)alloc(NHC * 2);
  unsigned short* g2   = (unsigned short*)alloc(NHC * 2);
  float*    proj = (float*)alloc(NHC * 4);

  // zero / -inf(encoded as 0) init, every call (deterministic)
  hipMemsetAsync(m1, 0, NH * 4, stream);
  hipMemsetAsync(m2, 0, NH * 4, stream);
  hipMemsetAsync(ma, 0, NH * 4, stream);
  hipMemsetAsync(z1, 0, NH * 4, stream);
  hipMemsetAsync(z2, 0, NH * 4, stream);
  hipMemsetAsync(za, 0, NH * 4, stream);
  hipMemsetAsync(hagg, 0, NHC * 4, stream);
  hipMemsetAsync(hact, 0, NHC * 4, stream);

  // bf16 conversions
  k_cvt_bf16<<<25000, 256, 0, stream>>>(x, xbf, 128, 128, 0, (long long)Nn * 128);
  auto cvtw = [&](const float* s, unsigned short* d, int ld, int o) {
    k_cvt_bf16<<<64, 256, 0, stream>>>(s, d, 128, ld, o, 16384);
  };
  cvtw(Wq1,  wB[0], 128, 0);  cvtw(Wq2,  wB[1], 128, 0);
  cvtw(Wk1,  wB[2], 128, 0);  cvtw(Wk2,  wB[3], 128, 0);
  cvtw(Wvd,  wB[4], 128, 0);  cvtw(Wqa,  wB[5], 128, 0);
  cvtw(Wka,  wB[6], 128, 0);  cvtw(Wva,  wB[7], 128, 0);
  cvtw(Win,  wB[8], 128, 0);  cvtw(Woutd,wB[9], 128, 0);
  cvtw(Wouta,wB[10],128, 0);
  cvtw(Wproj,wB[11],256, 0);  cvtw(Wproj,wB[12],256, 128);  // Wp1 / Wp2 split of [128,256]

  // node-side projections through WMMA (M = 50000 -> 3125 tiles)
  auto gemm = [&](const unsigned short* A, const unsigned short* B, const float* Cacc,
                  const float* bias, float* Df, unsigned short* Db) {
    k_wmma_gemm128<<<Nn / 16, 256, 0, stream>>>(A, B, Cacc, bias, Df, Db);
  };
  gemm(xbf, wB[0], nullptr, nullptr, nullptr, q1p);
  gemm(xbf, wB[1], nullptr, nullptr, nullptr, q2p);
  gemm(xbf, wB[2], nullptr, nullptr, nullptr, k1p);
  gemm(xbf, wB[3], nullptr, nullptr, nullptr, k2p);
  gemm(xbf, wB[4], nullptr, nullptr, nullptr, vdp);
  gemm(xbf, wB[5], nullptr, nullptr, nullptr, qap);
  gemm(xbf, wB[6], nullptr, nullptr, nullptr, kap);
  gemm(xbf, wB[7], nullptr, nullptr, nullptr, vap);
  gemm(xbf, wB[8], nullptr, b_in,    xinp,    nullptr);     // x@Win.T + b_in

  // edge pipeline (E*H = 3.2M threads -> 12500 blocks)
  k_edge_scores<<<12500, 256, 0, stream>>>(ei, ea, q1p, k1p, q2p, k2p, qap, kap,
                                           Wkr1, Wkr2, Wkra, t1, t2, ta, m1, m2, ma);
  k_edge_exp<<<12500, 256, 0, stream>>>(ei, t1, t2, ta, m1, m2, ma, z1, z2, za);
  k_edge_msg<<<12500, 256, 0, stream>>>(ei, ea, vdp, vap, Wvrd, Wvra,
                                        t1, t2, ta, z1, z2, za, lam, hagg, hact);

  // output projections + hybrid fusion (all WMMA)
  k_cvt_bf16<<<25000, 256, 0, stream>>>(hagg, habf, 128, 128, 0, (long long)NHC);
  k_cvt_bf16<<<25000, 256, 0, stream>>>(hact, hvbf, 128, 128, 0, (long long)NHC);
  gemm(habf, wB[9],  nullptr, nullptr, nullptr, g1);        // h_aggr @ Woutd.T
  gemm(hvbf, wB[10], nullptr, nullptr, nullptr, g2);        // h_actv @ Wouta.T
  gemm(g1,   wB[11], nullptr, bproj,   proj,    nullptr);   // g1 @ Wp1.T + bproj
  gemm(g2,   wB[12], proj,    nullptr, proj,    nullptr);   // += g2 @ Wp2.T

  // residual + RMSNorm -> d_out
  k_fuse<<<Nn, 128, 0, stream>>>(proj, xinp, rmsw, out);
}
